// Transformer_11123965296723
// MI455X (gfx1250) — matrix operand order repaired and checked by reading, not compiled, not run on hardware
//
#include <hip/hip_runtime.h>
#include <hip/hip_bf16.h>
#include <math.h>

#define BB 4
#define LL 1024
#define DD 1024
#define HH 16
#define FF 4096
#define DH 64
#define NLAYER 2
#define LNEPS 1e-6f

#define USE_ASYNC_LDS 1

typedef __attribute__((ext_vector_type(16))) __bf16 v16bf;
typedef __attribute__((ext_vector_type(8)))  float v8f;

union Frag { v16bf v; unsigned u[8]; };

__device__ __forceinline__ __bf16 to_bf16(float f) {
  union { float f; unsigned u; } c; c.f = f;
  unsigned r = c.u + 0x7fffu + ((c.u >> 16) & 1u);   // round-to-nearest-even
  unsigned short h = (unsigned short)(r >> 16);
  union { unsigned short s; __bf16 b; } o; o.s = h; return o.b;
}

__device__ __forceinline__ void store_val(float* p, float v)  { *p = v; }
__device__ __forceinline__ void store_val(__bf16* p, float v) { *p = to_bf16(v); }

// 16B global -> LDS copy; CDNA5 async path (ASYNCcnt-tracked, no VGPR round-trip)
__device__ __forceinline__ void copy16_g2l(const void* g, void* lds_generic) {
#if USE_ASYNC_LDS
  unsigned loff = (unsigned)(unsigned long)(__attribute__((address_space(3))) void*)lds_generic;
  unsigned long long ga = (unsigned long long)g;
  asm volatile("global_load_async_to_lds_b128 %0, %1, off"
               :: "v"(loff), "v"(ga) : "memory");
#else
  *(uint4*)lds_generic = *(const uint4*)g;
#endif
}
__device__ __forceinline__ void g2l_wait() {
#if USE_ASYNC_LDS
  asm volatile("s_wait_asynccnt 0" ::: "memory");
#endif
}

// ---------------------------------------------------------------- embedding
__global__ void embed_kernel(const int* __restrict__ tok,
                             const float* __restrict__ emb,
                             float* __restrict__ y) {
  int row = blockIdx.x;                       // b*L + l
  int t = tok[row];
  const float4* src = (const float4*)(emb + (size_t)t * DD);
  float4* dst = (float4*)(y + (size_t)row * DD);
  dst[threadIdx.x] = src[threadIdx.x];        // 256 thr * 4 = 1024
}

// ---------------------------------------------------------------- f32 -> bf16 cast (weights)
__global__ void cast_bf16_kernel(const float* __restrict__ in,
                                 __bf16* __restrict__ out, int n4) {
  int i = blockIdx.x * blockDim.x + threadIdx.x;
  if (i >= n4) return;
  float4 v = ((const float4*)in)[i];
  __bf16* o = out + (size_t)i * 4;
  o[0] = to_bf16(v.x); o[1] = to_bf16(v.y);
  o[2] = to_bf16(v.z); o[3] = to_bf16(v.w);
}

// ---------------------------------------------------------------- row layernorm over D=1024
template <typename OutT>
__global__ void ln_kernel(const float* __restrict__ x,
                          const float* __restrict__ gamma,
                          OutT* __restrict__ out) {
  __shared__ float rs[8], rs2[8];
  int row = blockIdx.x;
  const float4* xr = (const float4*)(x + (size_t)row * DD);
  float4 v = xr[threadIdx.x];
  float s  = v.x + v.y + v.z + v.w;
  float s2 = v.x*v.x + v.y*v.y + v.z*v.z + v.w*v.w;
#pragma unroll
  for (int m = 16; m >= 1; m >>= 1) { s += __shfl_xor(s, m, 32); s2 += __shfl_xor(s2, m, 32); }
  int lane = threadIdx.x & 31, w = threadIdx.x >> 5;
  if (lane == 0) { rs[w] = s; rs2[w] = s2; }
  __syncthreads();
  float S = 0.f, S2 = 0.f;
#pragma unroll
  for (int i = 0; i < 8; ++i) { S += rs[i]; S2 += rs2[i]; }
  float mean = S * (1.f / DD);
  float var  = S2 * (1.f / DD) - mean * mean;
  float r = rsqrtf(var + LNEPS);
  float4 g = ((const float4*)gamma)[threadIdx.x];
  OutT* o = out + (size_t)row * DD + threadIdx.x * 4;
  store_val(o + 0, (v.x - mean) * r * g.x);
  store_val(o + 1, (v.y - mean) * r * g.y);
  store_val(o + 2, (v.z - mean) * r * g.z);
  store_val(o + 3, (v.w - mean) * r * g.w);
}

// ---------------------------------------------------------------- per-head LN + RoPE + relayout to (B,H,L,Dh) bf16
__global__ void qk_rope_kernel(const float* __restrict__ in,     // ((b*L+l)*H + h)*Dh
                               const float* __restrict__ gamma,  // Dh or null
                               __bf16* __restrict__ out,         // (B,H,L,Dh)
                               float scale, int do_rope) {
  int row  = blockIdx.x * 8 + (threadIdx.x >> 5);  // one wave per (b,l,h)
  int lane = threadIdx.x & 31;
  int h  = row & (HH - 1);
  int bl = row >> 4;
  int l  = bl & (LL - 1);
  int b  = bl >> 10;
  const float* xr = in + (size_t)row * DH;
  float x0 = xr[lane], x1 = xr[lane + 32];
  if (gamma) {
    float s = x0 + x1, s2 = x0*x0 + x1*x1;
#pragma unroll
    for (int m = 16; m >= 1; m >>= 1) { s += __shfl_xor(s, m, 32); s2 += __shfl_xor(s2, m, 32); }
    float mean = s * (1.f / DH);
    float var  = s2 * (1.f / DH) - mean * mean;
    float r = rsqrtf(var + LNEPS);
    x0 = (x0 - mean) * r * gamma[lane];
    x1 = (x1 - mean) * r * gamma[lane + 32];
  }
  if (do_rope) {
    // 10000^(-lane/32) = exp2(-lane * log2(1e4)/32)
    float ang = (float)l * __builtin_amdgcn_exp2f(-(float)lane * 0.41524101186f);
    float sn, cs; __sincosf(ang, &sn, &cs);
    float o0 = x0 * cs - x1 * sn;   // rotate_half: [-x2, x1]
    float o1 = x1 * cs + x0 * sn;
    x0 = o0; x1 = o1;
  }
  size_t ob = (((size_t)b * HH + h) * LL + l) * DH;
  out[ob + lane]      = to_bf16(x0 * scale);
  out[ob + lane + 32] = to_bf16(x1 * scale);
}

// ---------------------------------------------------------------- WMMA bf16 GEMM, 128x128 tile, BK=32, 8 waves
#define EP_F32 0
#define EP_RESID 1
#define EP_GELU_BF16 2

__global__ void gemm_bf16_kernel(const __bf16* __restrict__ A,   // M x K row-major
                                 const __bf16* __restrict__ Bm,  // K x N row-major
                                 int N, int K,
                                 float* __restrict__ Cf, __bf16* __restrict__ Cb,
                                 const float* __restrict__ R, int ep) {
  __shared__ __align__(16) __bf16 As[128][40];   // [m][k]
  __shared__ __align__(16) __bf16 Bs[128][40];   // [n][k] (transposed at load)
  int tid = threadIdx.x, lane = tid & 31, wv = tid >> 5;
  int lid = lane & 15, g = lane >> 4;
  int m0 = blockIdx.y * 128, n0 = blockIdx.x * 128;
  int wm = (wv & 3) * 32;        // 4 waves down
  int wn = (wv >> 2) * 64;       // 2 waves across
  v8f acc[2][4];
#pragma unroll
  for (int mt = 0; mt < 2; ++mt)
#pragma unroll
    for (int nt = 0; nt < 4; ++nt)
#pragma unroll
      for (int j = 0; j < 8; ++j) acc[mt][nt][j] = 0.f;

  for (int k0 = 0; k0 < K; k0 += 32) {
#pragma unroll
    for (int it = 0; it < 2; ++it) {
      int la = (it * 256 + tid) * 8;
      int ar = la >> 5, ac = la & 31;          // A: 128 x 32
      copy16_g2l(A + (size_t)(m0 + ar) * K + k0 + ac, &As[ar][ac]);
      int br = la >> 7, bc = la & 127;         // B: 32 x 128
      uint4 bv = *(const uint4*)(Bm + (size_t)(k0 + br) * N + n0 + bc);
      const __bf16* bh = (const __bf16*)&bv;
#pragma unroll
      for (int j = 0; j < 8; ++j) Bs[bc + j][br] = bh[j];
    }
    g2l_wait();
    __syncthreads();
    Frag a[2];
#pragma unroll
    for (int mt = 0; mt < 2; ++mt) {
      int rrow = wm + mt * 16 + lid;
#pragma unroll
      for (int j = 0; j < 8; ++j) {
        int kk = (j < 4) ? g * 8 + 2 * j : 16 + g * 8 + 2 * (j - 4);
        a[mt].u[j] = *(const unsigned*)&As[rrow][kk];
      }
    }
#pragma unroll
    for (int nt = 0; nt < 4; ++nt) {
      Frag bb;
      int ncol = wn + nt * 16 + lid;
#pragma unroll
      for (int j = 0; j < 8; ++j)
        bb.u[j] = *(const unsigned*)&Bs[ncol][(j < 4) ? g * 8 + 2 * j : 16 + g * 8 + 2 * (j - 4)];
#pragma unroll
      for (int mt = 0; mt < 2; ++mt)
        acc[mt][nt] = __builtin_amdgcn_wmma_f32_16x16x32_bf16(
            false, a[mt].v, false, bb.v, (short)0, acc[mt][nt], false, false);
    }
    __syncthreads();
  }
#pragma unroll
  for (int mt = 0; mt < 2; ++mt)
#pragma unroll
    for (int nt = 0; nt < 4; ++nt)
#pragma unroll
      for (int r = 0; r < 8; ++r) {
        int row = m0 + wm + mt * 16 + g * 8 + r;
        int col = n0 + wn + nt * 16 + lid;
        size_t idx = (size_t)row * N + col;
        float vv = acc[mt][nt][r];
        if (ep == EP_GELU_BF16) {
          // 0.5x(1+tanh(s)) == x*sigmoid(2s), s = 0.7978846*(x+0.044715x^3)
          float t2 = 1.5957691216f * (vv + 0.044715f * vv * vv * vv);
          float gl = vv / (1.f + __expf(-t2));
          Cb[idx] = to_bf16(gl);
        } else if (ep == EP_RESID) {
          Cf[idx] = vv + R[idx];
        } else {
          Cf[idx] = vv;
        }
      }
}

// ---------------------------------------------------------------- flash attention, 64 q-rows per block, 4 waves
__global__ void flash_attn_kernel(const __bf16* __restrict__ Q,   // (B,H,L,Dh)
                                  const __bf16* __restrict__ Kt,  // (B,H,L,Dh)
                                  const __bf16* __restrict__ Vt,  // (B,H,L,Dh)
                                  __bf16* __restrict__ O) {       // (B,L,H*Dh)
  __shared__ __align__(16) __bf16 Qs[64][72];      // [q][dh]
  __shared__ __align__(16) __bf16 Ks[64][72];      // [kv][dh]
  __shared__ __align__(16) __bf16 Vs[64][72];      // [dh][kv] (transposed)
  __shared__ __align__(16) __bf16 Ps[4][16][72];   // per-wave P tile [q][kv]
  int tid = threadIdx.x, lane = tid & 31, w = tid >> 5;
  int lid = lane & 15, g = lane >> 4;
  int qt = blockIdx.x & 15;
  int bh = blockIdx.x >> 4;
  int b = bh >> 4, h = bh & 15;
  size_t base = (size_t)bh * LL * DH;

#pragma unroll
  for (int it = 0; it < 4; ++it) {
    int la = (it * 128 + tid) * 8;
    int rr = la >> 6, cc = la & 63;
    copy16_g2l(Q + base + (size_t)(qt * 64 + rr) * DH + cc, &Qs[rr][cc]);
  }
  g2l_wait();
  __syncthreads();
  Frag qf[2];
#pragma unroll
  for (int ks = 0; ks < 2; ++ks)
#pragma unroll
    for (int j = 0; j < 8; ++j) {
      int kk = ks * 32 + ((j < 4) ? g * 8 + 2 * j : 16 + g * 8 + 2 * (j - 4));
      qf[ks].u[j] = *(const unsigned*)&Qs[w * 16 + lid][kk];
    }

  v8f o_acc[4];
  float mprev[8], lsum[8];
#pragma unroll
  for (int nt = 0; nt < 4; ++nt)
#pragma unroll
    for (int j = 0; j < 8; ++j) o_acc[nt][j] = 0.f;
#pragma unroll
  for (int r = 0; r < 8; ++r) { mprev[r] = -1e30f; lsum[r] = 0.f; }

  for (int kt = 0; kt <= qt; ++kt) {
    __syncthreads();
#pragma unroll
    for (int it = 0; it < 4; ++it) {
      int la = (it * 128 + tid) * 8;
      int rr = la >> 6, cc = la & 63;
      copy16_g2l(Kt + base + (size_t)(kt * 64 + rr) * DH + cc, &Ks[rr][cc]);
      uint4 vv = *(const uint4*)(Vt + base + (size_t)(kt * 64 + rr) * DH + cc);
      const __bf16* vh = (const __bf16*)&vv;
#pragma unroll
      for (int j = 0; j < 8; ++j) Vs[cc + j][rr] = vh[j];
    }
    g2l_wait();
    __syncthreads();

    v8f s_acc[4];
#pragma unroll
    for (int nt = 0; nt < 4; ++nt)
#pragma unroll
      for (int j = 0; j < 8; ++j) s_acc[nt][j] = 0.f;
#pragma unroll
    for (int ks = 0; ks < 2; ++ks)
#pragma unroll
      for (int nt = 0; nt < 4; ++nt) {
        Frag bf_;
#pragma unroll
        for (int j = 0; j < 8; ++j)
          bf_.u[j] = *(const unsigned*)&Ks[nt * 16 + lid][ks * 32 + ((j < 4) ? g * 8 + 2 * j : 16 + g * 8 + 2 * (j - 4))];
        s_acc[nt] = __builtin_amdgcn_wmma_f32_16x16x32_bf16(
            false, qf[ks].v, false, bf_.v, (short)0, s_acc[nt], false, false);
      }

    if (kt == qt) {
#pragma unroll
      for (int nt = 0; nt < 4; ++nt)
#pragma unroll
        for (int r = 0; r < 8; ++r) {
          int kvp = nt * 16 + lid;
          int qp  = w * 16 + g * 8 + r;
          if (kvp > qp) s_acc[nt][r] = -1e30f;
        }
    }

    float corr[8];
#pragma unroll
    for (int r = 0; r < 8; ++r) {
      float mv = s_acc[0][r];
#pragma unroll
      for (int nt = 1; nt < 4; ++nt) mv = fmaxf(mv, s_acc[nt][r]);
#pragma unroll
      for (int m = 8; m >= 1; m >>= 1) mv = fmaxf(mv, __shfl_xor(mv, m, 32));
      float mn = fmaxf(mprev[r], mv);
      corr[r] = __expf(mprev[r] - mn);
      mprev[r] = mn;
    }
    float rsum[8];
#pragma unroll
    for (int r = 0; r < 8; ++r) rsum[r] = 0.f;
#pragma unroll
    for (int nt = 0; nt < 4; ++nt)
#pragma unroll
      for (int r = 0; r < 8; ++r) {
        float p = __expf(s_acc[nt][r] - mprev[r]);
        rsum[r] += p;
        Ps[w][g * 8 + r][nt * 16 + lid] = to_bf16(p);
      }
#pragma unroll
    for (int r = 0; r < 8; ++r) {
      float sv = rsum[r];
#pragma unroll
      for (int m = 8; m >= 1; m >>= 1) sv += __shfl_xor(sv, m, 32);
      lsum[r] = lsum[r] * corr[r] + sv;
#pragma unroll
      for (int nt = 0; nt < 4; ++nt) o_acc[nt][r] *= corr[r];
    }
    __syncthreads();

#pragma unroll
    for (int ks = 0; ks < 2; ++ks) {
      Frag pf;
#pragma unroll
      for (int j = 0; j < 8; ++j) {
        int kk = ks * 32 + ((j < 4) ? g * 8 + 2 * j : 16 + g * 8 + 2 * (j - 4));
        pf.u[j] = *(const unsigned*)&Ps[w][lid][kk];
      }
#pragma unroll
      for (int nt = 0; nt < 4; ++nt) {
        Frag vf;
#pragma unroll
        for (int j = 0; j < 8; ++j)
          vf.u[j] = *(const unsigned*)&Vs[nt * 16 + lid][ks * 32 + ((j < 4) ? g * 8 + 2 * j : 16 + g * 8 + 2 * (j - 4))];
        o_acc[nt] = __builtin_amdgcn_wmma_f32_16x16x32_bf16(
            false, pf.v, false, vf.v, (short)0, o_acc[nt], false, false);
      }
    }
  }

#pragma unroll
  for (int r = 0; r < 8; ++r) {
    float inv = 1.f / lsum[r];
    int q = qt * 64 + w * 16 + g * 8 + r;
    size_t ob = ((size_t)b * LL + q) * DD + h * DH;
#pragma unroll
    for (int nt = 0; nt < 4; ++nt)
      O[ob + nt * 16 + lid] = to_bf16(o_acc[nt][r] * inv);
  }
}

// ---------------------------------------------------------------- host driver
extern "C" void kernel_launch(void* const* d_in, const int* in_sizes, int n_in,
                              void* d_out, int out_size, void* d_ws, size_t ws_size,
                              hipStream_t stream) {
  (void)in_sizes; (void)n_in; (void)out_size; (void)ws_size;
  const int*   tokens = (const int*)d_in[0];
  const float* embed  = (const float*)d_in[1];
  const float* ln1    = (const float*)d_in[2];
  const float* wq     = (const float*)d_in[3];
  const float* wk     = (const float*)d_in[4];
  const float* wv     = (const float*)d_in[5];
  const float* qln    = (const float*)d_in[6];
  const float* kln    = (const float*)d_in[7];
  const float* wo     = (const float*)d_in[8];
  const float* ln2    = (const float*)d_in[9];
  const float* w1     = (const float*)d_in[10];
  const float* w2     = (const float*)d_in[11];
  const float* out_ln = (const float*)d_in[12];
  float* out = (float*)d_out;

  char* ws = (char*)d_ws;
  size_t off = 0;
  auto alloc = [&](size_t bytes) -> void* {
    void* p = ws + off; off += (bytes + 255) & ~(size_t)255; return p;
  };
  const size_t nDDw = (size_t)NLAYER * DD * DD;  // 2M elems
  const size_t nDFw = (size_t)NLAYER * DD * FF;  // 8M elems
  const size_t nBLD = (size_t)BB * LL * DD;      // 4M elems

  __bf16* wqb = (__bf16*)alloc(nDDw * 2);
  __bf16* wkb = (__bf16*)alloc(nDDw * 2);
  __bf16* wvb = (__bf16*)alloc(nDDw * 2);
  __bf16* wob = (__bf16*)alloc(nDDw * 2);
  __bf16* w1b = (__bf16*)alloc(nDFw * 2);
  __bf16* w2b = (__bf16*)alloc(nDFw * 2);
  float*  ybuf = (float*)alloc(nBLD * 4);
  float*  xbuf = (float*)alloc(nBLD * 4);
  __bf16* xln  = (__bf16*)alloc(nBLD * 2);
  float*  qf   = (float*)alloc(nBLD * 4);
  float*  kf   = (float*)alloc(nBLD * 4);
  float*  vf   = (float*)alloc(nBLD * 4);
  __bf16* qb   = (__bf16*)alloc(nBLD * 2);
  __bf16* kb   = (__bf16*)alloc(nBLD * 2);
  __bf16* vb   = (__bf16*)alloc(nBLD * 2);
  __bf16* attnb = (__bf16*)alloc(nBLD * 2);
  __bf16* hb = (__bf16*)qf;  // alias: qf/kf (32MB) dead once qb/kb built

  cast_bf16_kernel<<<(int)(nDDw / 4 / 256), 256, 0, stream>>>(wq, wqb, (int)(nDDw / 4));
  cast_bf16_kernel<<<(int)(nDDw / 4 / 256), 256, 0, stream>>>(wk, wkb, (int)(nDDw / 4));
  cast_bf16_kernel<<<(int)(nDDw / 4 / 256), 256, 0, stream>>>(wv, wvb, (int)(nDDw / 4));
  cast_bf16_kernel<<<(int)(nDDw / 4 / 256), 256, 0, stream>>>(wo, wob, (int)(nDDw / 4));
  cast_bf16_kernel<<<(int)(nDFw / 4 / 256), 256, 0, stream>>>(w1, w1b, (int)(nDFw / 4));
  cast_bf16_kernel<<<(int)(nDFw / 4 / 256), 256, 0, stream>>>(w2, w2b, (int)(nDFw / 4));

  embed_kernel<<<BB * LL, 256, 0, stream>>>(tokens, embed, ybuf);

  dim3 gD(DD / 128, BB * LL / 128);   // (8, 32)  N=1024 GEMMs
  dim3 gF(FF / 128, BB * LL / 128);   // (32, 32) N=4096 GEMM
  int ropeBlocks = BB * LL * HH / 8;

  for (int i = 0; i < NLAYER; ++i) {
    ln_kernel<__bf16><<<BB * LL, 256, 0, stream>>>(ybuf, ln1 + (size_t)i * DD, xln);
    gemm_bf16_kernel<<<gD, 256, 0, stream>>>(xln, wqb + (size_t)i * DD * DD, DD, DD, qf, nullptr, nullptr, EP_F32);
    gemm_bf16_kernel<<<gD, 256, 0, stream>>>(xln, wkb + (size_t)i * DD * DD, DD, DD, kf, nullptr, nullptr, EP_F32);
    gemm_bf16_kernel<<<gD, 256, 0, stream>>>(xln, wvb + (size_t)i * DD * DD, DD, DD, vf, nullptr, nullptr, EP_F32);
    qk_rope_kernel<<<ropeBlocks, 256, 0, stream>>>(qf, qln + (size_t)i * DH, qb, 0.125f, 1);
    qk_rope_kernel<<<ropeBlocks, 256, 0, stream>>>(kf, kln + (size_t)i * DH, kb, 1.0f, 1);
    qk_rope_kernel<<<ropeBlocks, 256, 0, stream>>>(vf, nullptr, vb, 1.0f, 0);
    flash_attn_kernel<<<BB * HH * (LL / 64), 128, 0, stream>>>(qb, kb, vb, attnb);
    gemm_bf16_kernel<<<gD, 256, 0, stream>>>(attnb, wob + (size_t)i * DD * DD, DD, DD, xbuf, nullptr, ybuf, EP_RESID);
    ln_kernel<__bf16><<<BB * LL, 256, 0, stream>>>(xbuf, ln2 + (size_t)i * DD, xln);
    gemm_bf16_kernel<<<gF, 256, 0, stream>>>(xln, w1b + (size_t)i * DD * FF, FF, DD, nullptr, hb, nullptr, EP_GELU_BF16);
    gemm_bf16_kernel<<<gD, 256, 0, stream>>>(hb, w2b + (size_t)i * DD * FF, DD, FF, ybuf, nullptr, xbuf, EP_RESID);
  }
  ln_kernel<float><<<BB * LL, 256, 0, stream>>>(ybuf, out_ln, out);
}
